// PerceiverAlibiSelfAttention_66718021976280
// MI455X (gfx1250) — matrix operand order repaired and checked by reading, not compiled, not run on hardware
//
#include <hip/hip_runtime.h>
#include <math.h>
#include <stdint.h>

typedef __attribute__((ext_vector_type(16))) _Float16 v16h;
typedef __attribute__((ext_vector_type(8)))  _Float16 v8h;
typedef __attribute__((ext_vector_type(8)))  float    v8f;

#define B_  4
#define S_  2048
#define D_  1024
#define H_  16
#define DH_ 64
#define LN_EPS 1e-5f

// gfx1250 async global->LDS copy path (ASYNCcnt-tracked). Set to 0 if the
// assembler rejects the mnemonic/operands.
#define USE_ASYNC_LDS 1

#define WMMA_F16(a, b, c) \
  __builtin_amdgcn_wmma_f32_16x16x32_f16(false, (a), false, (b), (short)0, (c), false, false)

// Build a 16x32 f16 A-fragment from its two contiguous 16B half-rows.
static __device__ __forceinline__ v16h load_a16(const _Float16* lo, const _Float16* hi) {
  union { v16h v; struct { v8h lo, hi; } s; } u;
  u.s.lo = *(const v8h*)lo;
  u.s.hi = *(const v8h*)hi;
  return u.v;
}

// Copy 16 bytes global -> LDS. Async path: per-lane GLOBAL_LOAD_ASYNC_TO_LDS_B128
// (VDST = LDS byte offset = low 32 bits of flat shared address per ISA aperture rule).
static __device__ __forceinline__ void cp16(_Float16* lds_dst, const _Float16* gsrc) {
#if USE_ASYNC_LDS
  uint32_t lds_off = (uint32_t)(uintptr_t)(void*)lds_dst;
  asm volatile("global_load_async_to_lds_b128 %0, %1, off"
               :: "v"(lds_off), "v"(gsrc) : "memory");
#else
  *(v8h*)lds_dst = *(const v8h*)gsrc;
#endif
}

static __device__ __forceinline__ void async_copy_wait() {
#if USE_ASYNC_LDS
  asm volatile("s_wait_asynccnt 0" ::: "memory");
#endif
}

// ---------------- Kernel 1: LayerNorm (one block = one row of 1024) ----------
__global__ void ln_kernel(const float* __restrict__ x,
                          const float* __restrict__ gamma,
                          const float* __restrict__ beta,
                          _Float16* __restrict__ xh) {
  __shared__ float red[256];
  const int row = blockIdx.x;
  const int tid = threadIdx.x;
  const float* xr = x + (size_t)row * D_;
  float v0 = xr[tid], v1 = xr[tid + 256], v2 = xr[tid + 512], v3 = xr[tid + 768];
  red[tid] = v0 + v1 + v2 + v3;
  __syncthreads();
  for (int off = 128; off > 0; off >>= 1) {
    if (tid < off) red[tid] += red[tid + off];
    __syncthreads();
  }
  float mu = red[0] * (1.0f / D_);
  __syncthreads();
  float d0 = v0 - mu, d1 = v1 - mu, d2 = v2 - mu, d3 = v3 - mu;
  red[tid] = d0 * d0 + d1 * d1 + d2 * d2 + d3 * d3;
  __syncthreads();
  for (int off = 128; off > 0; off >>= 1) {
    if (tid < off) red[tid] += red[tid + off];
    __syncthreads();
  }
  float rstd = rsqrtf(red[0] * (1.0f / D_) + LN_EPS);
  _Float16* o = xh + (size_t)row * D_;
  o[tid]       = (_Float16)(d0 * rstd * gamma[tid]       + beta[tid]);
  o[tid + 256] = (_Float16)(d1 * rstd * gamma[tid + 256] + beta[tid + 256]);
  o[tid + 512] = (_Float16)(d2 * rstd * gamma[tid + 512] + beta[tid + 512]);
  o[tid + 768] = (_Float16)(d3 * rstd * gamma[tid + 768] + beta[tid + 768]);
}

// -------- Kernel 2: convert weights to f16, transposed: Wt[w][n][k] = W[k][n] --
__global__ void wconv_kernel(const float* __restrict__ Wq,
                             const float* __restrict__ Wk,
                             const float* __restrict__ Wv,
                             _Float16* __restrict__ Wt) {
  int idx = blockIdx.x * blockDim.x + threadIdx.x;   // [0, 3*1024*1024)
  int w   = idx >> 20;
  int rem = idx & 1048575;
  int n   = rem >> 10;
  int kk  = rem & 1023;
  const float* W = (w == 0) ? Wq : ((w == 1) ? Wk : Wv);
  Wt[idx] = (_Float16)W[kk * 1024 + n];
}

// -------- Kernel 3: QKV projection GEMM (one wave per 32x64 tile) -------------
// grid = (M/32=256, N/64=16, 3), block = 32 (one wave)
// 8 WMMAs per K-step on 2 A-frags x 4 B-frags -> ~24B of loads per WMMA.
__global__ void qkv_gemm_kernel(const _Float16* __restrict__ xh,
                                const _Float16* __restrict__ Wt,
                                const float* __restrict__ bq,
                                const float* __restrict__ bk,
                                const float* __restrict__ bv,
                                _Float16* __restrict__ qh,
                                _Float16* __restrict__ kh,
                                _Float16* __restrict__ vT) {
  const int lane   = threadIdx.x & 31;
  const int ln15   = lane & 15;
  const int hiHalf = lane >> 4;          // 0 or 1
  const int kbase8  = hiHalf * 8;
  const int kbase16 = hiHalf * 16;
  const int w  = blockIdx.z;
  const int m0 = blockIdx.x * 32;
  const int n0 = blockIdx.y * 64;

  const _Float16* Arow0 = xh + (size_t)(m0 + ln15) * D_;
  const _Float16* Arow1 = Arow0 + (size_t)16 * D_;
  const _Float16* Brow  = Wt + ((size_t)w << 20) + (size_t)(n0 + ln15) * D_;

  v8f acc[2][4] = {};
  for (int kk = 0; kk < D_; kk += 32) {
    v16h a0 = load_a16(Arow0 + kk + kbase16, Arow0 + kk + kbase16 + 8);
    v16h a1 = load_a16(Arow1 + kk + kbase16, Arow1 + kk + kbase16 + 8);
#pragma unroll
    for (int t = 0; t < 4; ++t) {
      v16h b = *(const v16h*)(Brow + (size_t)t * 16 * D_ + kk + kbase16);
      acc[0][t] = WMMA_F16(a0, b, acc[0][t]);
      acc[1][t] = WMMA_F16(a1, b, acc[1][t]);
    }
  }

  const float* bias = (w == 0) ? bq : ((w == 1) ? bk : bv);
#pragma unroll
  for (int t = 0; t < 4; ++t) {
    const int ncol = n0 + t * 16 + ln15;
    const float bb = bias[ncol];
    const int h = ncol >> 6, d = ncol & 63;
#pragma unroll
    for (int mt = 0; mt < 2; ++mt) {
#pragma unroll
      for (int r = 0; r < 8; ++r) {
        int mrow = m0 + mt * 16 + r + 8 * hiHalf;
        int bidx = mrow >> 11, s = mrow & 2047;
        float val = acc[mt][t][r] + bb;
        if (w == 2) {
          // V stored transposed per head: [B][H][DH][S]
          vT[((size_t)(bidx * H_ + h) * DH_ + d) * S_ + s] = (_Float16)val;
        } else {
          _Float16* dst = (w == 0) ? qh : kh;      // [B][H][S][DH]
          dst[((size_t)(bidx * H_ + h) * S_ + s) * DH_ + d] = (_Float16)val;
        }
      }
    }
  }
}

// -------- Kernel 4: flash attention with ALiBi (4 waves, 16 q-rows/wave) ------
// grid = B*H*(S/64) = 2048, block = 128.
// K/V tiles staged once per block into LDS via async global->LDS copies.
__global__ void attn_kernel(const _Float16* __restrict__ qh,
                            const _Float16* __restrict__ kh,
                            const _Float16* __restrict__ vT,
                            float* __restrict__ out) {
  __shared__ __align__(64) _Float16 Kl[32 * 64];      // 32 keys x 64 dh (row-major)
  __shared__ __align__(64) _Float16 Vl[64 * 32];      // 64 dh x 32 keys (row-major)
  __shared__ __align__(64) _Float16 plds[4][16 * 32]; // per-wave P tiles
  const int tid    = threadIdx.x;
  const int lane   = tid & 31;
  const int wv     = tid >> 5;
  const int ln15   = lane & 15;
  const int hiHalf = lane >> 4;
  const int kbase8  = hiHalf * 8;
  const int kbase16 = hiHalf * 16;

  int blk = blockIdx.x;              // (b*16 + h)*32 + qb
  const int qb = blk & 31; blk >>= 5;
  const int h  = blk & 15;
  const int b  = blk >> 4;
  const int q0 = qb * 64 + wv * 16;

  const size_t head = (size_t)(b * H_ + h) * S_ * DH_;
  const _Float16* Q = qh + head;     // [S][DH]
  const _Float16* K = kh + head;     // [S][DH]
  const _Float16* V = vT + head;     // [DH][S]

  // Persistent Q A-fragments (DH=64 -> two K=32 chunks)
  const _Float16* Qrow = Q + (size_t)(q0 + ln15) * DH_;
  const v16h qf0 = load_a16(Qrow + 0  + kbase16, Qrow + 8  + kbase16);
  const v16h qf1 = load_a16(Qrow + 32 + kbase16, Qrow + 40 + kbase16);

  const float slope = exp2f(-0.5f * (float)(h + 1));   // ALiBi slope for H=16
  const float qi    = (float)(q0 + 8 * hiHalf);

  float mrun[8], lrun[8];
  v8f o0 = {}, o1 = {}, o2 = {}, o3 = {};
#pragma unroll
  for (int r = 0; r < 8; ++r) { mrun[r] = -__builtin_inff(); lrun[r] = 0.0f; }

  _Float16* P = &plds[wv][0];

  // V-tile copy assignments (64 rows x 4 chunks of 8 halves = 256 chunks, 2/thread)
  const int vc0 = tid, vc1 = tid + 128;
  const int vr0 = vc0 >> 2, vp0 = vc0 & 3;
  const int vr1 = vc1 >> 2, vp1 = vc1 & 3;

  for (int j0 = 0; j0 < S_; j0 += 32) {
    __syncthreads();   // previous iteration's LDS reads complete
    // K tile: 4KB contiguous in global (rows j0..j0+31, 64 halves each)
    {
      const _Float16* src = K + (size_t)j0 * DH_;
      cp16(Kl + tid * 8,         src + tid * 8);
      cp16(Kl + (tid + 128) * 8, src + (tid + 128) * 8);
      cp16(Vl + vr0 * 32 + vp0 * 8, V + (size_t)vr0 * S_ + j0 + vp0 * 8);
      cp16(Vl + vr1 * 32 + vp1 * 8, V + (size_t)vr1 * S_ + j0 + vp1 * 8);
    }
    async_copy_wait();
    __syncthreads();   // tiles visible to all waves

    // ---- scores: two 16-key tiles, K-dim = DH in two chunks of 32 ----
    v16h kb00 = *(const v16h*)(Kl + (ln15)      * 64 + 0  + kbase16);
    v16h kb01 = *(const v16h*)(Kl + (ln15)      * 64 + 32 + kbase16);
    v16h kb10 = *(const v16h*)(Kl + (16 + ln15) * 64 + 0  + kbase16);
    v16h kb11 = *(const v16h*)(Kl + (16 + ln15) * 64 + 32 + kbase16);
    v8f s0 = {}, s1 = {};
    s0 = WMMA_F16(qf0, kb00, s0);
    s0 = WMMA_F16(qf1, kb01, s0);
    s1 = WMMA_F16(qf0, kb10, s1);
    s1 = WMMA_F16(qf1, kb11, s1);

    // ---- ALiBi bias + scale + online softmax (rows live in 16-lane groups) ----
    float p0[8], p1[8], corr[8];
#pragma unroll
    for (int r = 0; r < 8; ++r) {
      float i  = qi + (float)r;
      float ja = (float)(j0 + ln15);
      float jb = (float)(j0 + 16 + ln15);
      float sa = (s0[r] - slope * fabsf(ja - i)) * 0.125f;   // (qk + bias)/sqrt(64)
      float sb = (s1[r] - slope * fabsf(jb - i)) * 0.125f;
      float rm = fmaxf(sa, sb);
      rm = fmaxf(rm, __shfl_xor(rm, 1, 16));
      rm = fmaxf(rm, __shfl_xor(rm, 2, 16));
      rm = fmaxf(rm, __shfl_xor(rm, 4, 16));
      rm = fmaxf(rm, __shfl_xor(rm, 8, 16));
      float mnew = fmaxf(mrun[r], rm);
      corr[r] = __expf(mrun[r] - mnew);
      mrun[r] = mnew;
      p0[r] = __expf(sa - mnew);
      p1[r] = __expf(sb - mnew);
      lrun[r] = lrun[r] * corr[r] + p0[r] + p1[r];
    }
#pragma unroll
    for (int r = 0; r < 8; ++r) {
      o0[r] *= corr[r]; o1[r] *= corr[r]; o2[r] *= corr[r]; o3[r] *= corr[r];
    }

    // ---- P: C-layout -> A-fragment layout via per-wave LDS tile (16x32 f16) ----
#pragma unroll
    for (int r = 0; r < 8; ++r) {
      int m = r + 8 * hiHalf;
      P[m * 32 + ln15]      = (_Float16)p0[r];
      P[m * 32 + 16 + ln15] = (_Float16)p1[r];
    }
    asm volatile("s_wait_dscnt 0" ::: "memory");   // wave-local LDS store->load
    v16h pf = load_a16(P + ln15 * 32 + kbase16, P + ln15 * 32 + kbase16 + 8);

    // ---- O += P(16x32) x V(32x16) per DH chunk, V from LDS ----
    v16h vb;
    vb = *(const v16h*)(Vl + (0  + ln15) * 32 + kbase16);
    o0 = WMMA_F16(pf, vb, o0);
    vb = *(const v16h*)(Vl + (16 + ln15) * 32 + kbase16);
    o1 = WMMA_F16(pf, vb, o1);
    vb = *(const v16h*)(Vl + (32 + ln15) * 32 + kbase16);
    o2 = WMMA_F16(pf, vb, o2);
    vb = *(const v16h*)(Vl + (48 + ln15) * 32 + kbase16);
    o3 = WMMA_F16(pf, vb, o3);
  }

  // ---- normalize by row sums and store fp32 output [B][S][H*DH] ----
#pragma unroll
  for (int r = 0; r < 8; ++r) {
    float l = lrun[r];
    l += __shfl_xor(l, 1, 16);
    l += __shfl_xor(l, 2, 16);
    l += __shfl_xor(l, 4, 16);
    l += __shfl_xor(l, 8, 16);
    float inv = 1.0f / l;
    int m = r + 8 * hiHalf;
    size_t base = ((size_t)(b * S_ + q0 + m)) * D_ + h * DH_ + ln15;
    out[base +  0] = o0[r] * inv;
    out[base + 16] = o1[r] * inv;
    out[base + 32] = o2[r] * inv;
    out[base + 48] = o3[r] * inv;
  }
}

extern "C" void kernel_launch(void* const* d_in, const int* in_sizes, int n_in,
                              void* d_out, int out_size, void* d_ws, size_t ws_size,
                              hipStream_t stream) {
  const float* hidden = (const float*)d_in[0];
  const float* gamma  = (const float*)d_in[1];
  const float* beta   = (const float*)d_in[2];
  const float* Wq     = (const float*)d_in[3];
  const float* bq     = (const float*)d_in[4];
  const float* Wk     = (const float*)d_in[5];
  const float* bk     = (const float*)d_in[6];
  const float* Wv     = (const float*)d_in[7];
  const float* bv     = (const float*)d_in[8];
  float* out = (float*)d_out;

  char* ws = (char*)d_ws;
  _Float16* xh = (_Float16*)(ws);                          // 16 MiB: LN(x) f16
  _Float16* Wt = (_Float16*)(ws + (size_t)(16u << 20));    //  6 MiB: Wq/Wk/Wv f16, transposed
  _Float16* qh = (_Float16*)(ws + (size_t)(22u << 20));    // 16 MiB: Q [B][H][S][DH]
  _Float16* kh = (_Float16*)(ws + (size_t)(38u << 20));    // 16 MiB: K [B][H][S][DH]
  _Float16* vT = (_Float16*)(ws + (size_t)(54u << 20));    // 16 MiB: V [B][H][DH][S]

  ln_kernel<<<B_ * S_, 256, 0, stream>>>(hidden, gamma, beta, xh);
  wconv_kernel<<<(3 * 1024 * 1024) / 256, 256, 0, stream>>>(Wq, Wk, Wv, Wt);
  qkv_gemm_kernel<<<dim3(256, 16, 3), 32, 0, stream>>>(xh, Wt, bq, bk, bv, qh, kh, vT);
  attn_kernel<<<B_ * H_ * (S_ / 64), 128, 0, stream>>>(qh, kh, vT, out);
}
